// LstmModel_936302870531
// MI455X (gfx1250) — matrix operand order repaired and checked by reading, not compiled, not run on hardware
//
#include <hip/hip_runtime.h>

// ---------------- CDNA5 WMMA types ----------------
typedef __attribute__((ext_vector_type(16))) __bf16 v16bf;
typedef __attribute__((ext_vector_type(8)))  float  v8f;

typedef union {
    v16bf v;
    uint4 q[2];
} frag_u;

#define HH   50
#define DD   8
#define TT   512
#define GP   208      // padded gate dim (13 tiles of 16)
#define NT   13       // gate tiles
#define K0P  64       // layer0 K pad: [x(8) | h0(50) | 1(bias) | 0..]
#define K1P  128      // layer1 K pad: [h0(50) | h1(50) | 1(bias) | 0..]
#define NTHREADS 416  // 13 waves of 32

#define BIAS_K0 (DD + HH)      // 58: constant-1 column, weight row = bias0
#define BIAS_K1 (2 * HH)       // 100: constant-1 column, weight row = bias1

// packed weight sizes (bf16 elements)
#define W0_FRAGS (NT * 2)
#define W1_FRAGS (NT * 4)
#define FRAG_ELEMS 512                      // 32 lanes * 16 bf16
#define W0_ELEMS (W0_FRAGS * FRAG_ELEMS)    // 13312
#define WPACK_ELEMS ((W0_FRAGS + W1_FRAGS) * FRAG_ELEMS)  // 39936
#define WPACK_BYTES (WPACK_ELEMS * 2)       // 79872

// LDS layout (bytes)
#define SMEM_W   0
#define SMEM_A0  (WPACK_BYTES)             // 16*64  bf16 = 2048
#define SMEM_A1  (SMEM_A0 + 16*K0P*2)      // 16*128 bf16 = 4096
#define SMEM_G0  (SMEM_A1 + 16*K1P*2)      // 16*208 f32  = 13312
#define SMEM_G1  (SMEM_G0 + 16*GP*4)
#define SMEM_C0  (SMEM_G1 + 16*GP*4)       // 800 f32 = 3200
#define SMEM_C1  (SMEM_C0 + 16*HH*4)
#define SMEM_X   (SMEM_C1 + 16*HH*4)       // 16*8 f32 = 512 (async x staging)
#define SMEM_TOT (SMEM_X + 16*DD*4)        // 119552

// ---------------- weight pack kernel ----------------
// W = [Wih^T ; Whh^T ; bias-row ; zero-pad] (K x 208) packed into WMMA
// B-fragment lane order: fragment (tile, kchunk) = B-tile K=kchunk*32..+31,
// N=tile*16..+15; lane l -> column (l&15), K rows (l>>4)*16 + e (e ascending),
// 32 contiguous bytes per lane.
__global__ void pack_weights(const float* __restrict__ Wih0, const float* __restrict__ Whh0,
                             const float* __restrict__ bih0, const float* __restrict__ bhh0,
                             const float* __restrict__ Wih1, const float* __restrict__ Whh1,
                             const float* __restrict__ bih1, const float* __restrict__ bhh1,
                             __bf16* __restrict__ wpack)
{
    int i = blockIdx.x * blockDim.x + threadIdx.x;
    if (i >= WPACK_ELEMS) return;
    float val = 0.0f;
    if (i < W0_ELEMS) {
        int f    = i >> 9;
        int r    = i & 511;
        int tile = f >> 1, c = f & 1;
        int lane = r >> 4, e = r & 15;
        int col  = tile * 16 + (lane & 15);
        int k    = c * 32 + ((lane >> 4) << 4) + e;
        if (col < 4 * HH) {
            if (k < DD)             val = Wih0[col * DD + k];
            else if (k < DD + HH)   val = Whh0[col * HH + (k - DD)];
            else if (k == BIAS_K0)  val = bih0[col] + bhh0[col];
        }
    } else {
        int ii   = i - W0_ELEMS;
        int f    = ii >> 9;
        int r    = ii & 511;
        int tile = f >> 2, c = f & 3;
        int lane = r >> 4, e = r & 15;
        int col  = tile * 16 + (lane & 15);
        int k    = c * 32 + ((lane >> 4) << 4) + e;
        if (col < 4 * HH) {
            if (k < HH)             val = Wih1[col * HH + k];
            else if (k < 2 * HH)    val = Whh1[col * HH + (k - HH)];
            else if (k == BIAS_K1)  val = bih1[col] + bhh1[col];
        }
    }
    wpack[i] = (__bf16)val;
}

// ---------------- fragment loaders (from LDS) ----------------
__device__ __forceinline__ v16bf load_afrag(const __bf16* A, int stride, int kbase, int lane) {
    const __bf16* p = A + (lane & 15) * stride + kbase + ((lane >> 4) << 4);
    frag_u u;
    u.q[0] = *(const uint4*)(p);
    u.q[1] = *(const uint4*)(p + 8);
    return u.v;
}
__device__ __forceinline__ v16bf load_bfrag(const __bf16* base, int lane) {
    const __bf16* p = base + lane * 16;
    frag_u u;
    u.q[0] = *(const uint4*)(p);
    u.q[1] = *(const uint4*)(p + 8);
    return u.v;
}

// ---------------- fast activations (gfx1250 V_TANH_F32) ----------------
#if __has_builtin(__builtin_amdgcn_tanhf)
__device__ __forceinline__ float tanh_fast(float x) { return __builtin_amdgcn_tanhf(x); }
#else
__device__ __forceinline__ float tanh_fast(float x) {
    float e = __expf(2.0f * x);
    return (e - 1.0f) * __builtin_amdgcn_rcpf(e + 1.0f);
}
#endif
__device__ __forceinline__ float sigmoid_fast(float x) {
    return fmaf(0.5f, tanh_fast(0.5f * x), 0.5f);
}

// gates at g[0]=i, g[HH]=f, g[2HH]=g, g[3HH]=o (bias already folded in GEMM)
__device__ __forceinline__ float lstm_cell(const float* g, float cprev, float* cout) {
    float ig = sigmoid_fast(g[0]);
    float fg = sigmoid_fast(g[HH]);
    float gg = tanh_fast(g[2 * HH]);
    float og = sigmoid_fast(g[3 * HH]);
    float c  = fmaf(fg, cprev, ig * gg);
    *cout = c;
    return og * tanh_fast(c);
}

__device__ __forceinline__ void wait_async0() {
#if __has_builtin(__builtin_amdgcn_s_wait_asynccnt)
    __builtin_amdgcn_s_wait_asynccnt(0);
#else
    asm volatile("s_wait_asynccnt 0" ::: "memory");
#endif
}

// ---------------- fused 2-layer LSTM ----------------
// grid = B/16 blocks; 13 waves/block, wave w owns gate N-tile w for both layers.
__global__ __launch_bounds__(NTHREADS)
void lstm_fused(const float* __restrict__ x,
                const __bf16* __restrict__ wpack,
                const float* __restrict__ Wlin,
                const float* __restrict__ blin,
                float* __restrict__ out)
{
    extern __shared__ char smem[];
    __bf16* sW  = (__bf16*)(smem + SMEM_W);
    __bf16* sA0 = (__bf16*)(smem + SMEM_A0);   // [16][K0P] = [x_t | h0 | 1 | 0]
    __bf16* sA1 = (__bf16*)(smem + SMEM_A1);   // [16][K1P] = [h0_t | h1 | 1 | 0]
    float*  sG0 = (float*) (smem + SMEM_G0);   // [16][GP]
    float*  sG1 = (float*) (smem + SMEM_G1);
    float*  sC0 = (float*) (smem + SMEM_C0);   // [800]
    float*  sC1 = (float*) (smem + SMEM_C1);
    float*  sXf = (float*) (smem + SMEM_X);    // [16][8] async x staging (f32)

    const int tid  = threadIdx.x;
    const int wave = tid >> 5;     // 0..12 == gate tile index
    const int lane = tid & 31;
    const int row0 = blockIdx.x * 16;

    // ---- one-time setup ----
    for (int i = tid; i < WPACK_BYTES / 16; i += NTHREADS)
        ((uint4*)sW)[i] = ((const uint4*)wpack)[i];
    for (int i = tid; i < 16 * K0P; i += NTHREADS) sA0[i] = (__bf16)0.0f;
    for (int i = tid; i < 16 * K1P; i += NTHREADS) sA1[i] = (__bf16)0.0f;
    for (int i = tid; i < 16 * HH;  i += NTHREADS) { sC0[i] = 0.0f; sC1[i] = 0.0f; }
    if (tid < 16 * DD) {  // x_0 -> A0 cols 0..7 (cold direct load, once)
        int m = tid >> 3, d = tid & 7;
        sA0[m * K0P + d] = (__bf16)x[((size_t)(row0 + m) * TT + 0) * DD + d];
    }
    if (tid < 16) {       // constant-1 bias columns
        sA0[tid * K0P + BIAS_K0] = (__bf16)1.0f;
        sA1[tid * K1P + BIAS_K1] = (__bf16)1.0f;
    }

    // async x prefetch setup: wave0's 32 lanes fetch one 16-row x-step tile
    // (16 rows x 32B, one b128 per lane) into sXf, two steps ahead.
    const unsigned ldsa = (unsigned)(size_t)(smem + SMEM_X)
                        + ((lane >> 1) << 5) + ((lane & 1) << 4);
    const unsigned voff = (unsigned)((row0 + (lane >> 1)) * (TT * DD * 4))
                        + ((lane & 1) << 4);
    if (wave == 0) {      // prefetch x_1
        unsigned long long base = (unsigned long long)(const void*)x + 32ull;
        asm volatile("global_load_async_to_lds_b128 %0, %1, %2"
                     :: "v"(ldsa), "v"(voff), "s"(base) : "memory");
    }

    // precomputed pointwise indices/pointers (each thread owns <=2 of 800 cells)
    const int m0 = tid / HH, j0 = tid - m0 * HH;
    const int i1 = tid + NTHREADS;
    const bool has1 = (i1 < 16 * HH);
    const int m1 = i1 / HH, j1 = i1 - m1 * HH;
    const float* g0e0 = sG0 + m0 * GP + j0;
    const float* g0e1 = sG0 + m1 * GP + j1;
    const float* g1e0 = sG1 + m0 * GP + j0;
    const float* g1e1 = sG1 + m1 * GP + j1;
    __bf16* hA0e0 = sA0 + m0 * K0P + DD + j0;   // h0 recurrent slot
    __bf16* hA0e1 = sA0 + m1 * K0P + DD + j1;
    __bf16* hA1e0 = sA1 + m0 * K1P + j0;        // layer-1 input slot
    __bf16* hA1e1 = sA1 + m1 * K1P + j1;
    __bf16* hR1e0 = sA1 + m0 * K1P + HH + j0;   // h1 recurrent slot
    __bf16* hR1e1 = sA1 + m1 * K1P + HH + j1;

    __syncthreads();

    const int col = wave * 16 + (lane & 15);
    const int rb  = (lane >> 4) * 8;

    for (int t = 0; t < TT; ++t) {
        // ---- layer 0 GEMM: gates0 = [x_t|h0|1] @ W0 (2 k-chunks) ----
        {
            v8f acc = {};
#pragma unroll
            for (int c = 0; c < 2; ++c) {
                v16bf a = load_afrag(sA0, K0P, c * 32, lane);
                v16bf b = load_bfrag(sW + (size_t)(wave * 2 + c) * FRAG_ELEMS, lane);
                acc = __builtin_amdgcn_wmma_f32_16x16x32_bf16(
                        false, a, false, b, (short)0, acc, false, false);
            }
#pragma unroll
            for (int v = 0; v < 8; ++v) sG0[(rb + v) * GP + col] = acc[v];
        }
        if (wave == 0) wait_async0();   // x_{t+1} landed in sXf
        __syncthreads();                                        // (b)

        // ---- layer 0 pointwise + publish x_{t+1} into A0 ----
        {
            float h = lstm_cell(g0e0, sC0[tid], &sC0[tid]);
            __bf16 hb = (__bf16)h;
            *hA0e0 = hb; *hA1e0 = hb;
            if (has1) {
                float h2 = lstm_cell(g0e1, sC0[i1], &sC0[i1]);
                __bf16 hb2 = (__bf16)h2;
                *hA0e1 = hb2; *hA1e1 = hb2;
            }
        }
        if (t + 1 < TT && tid < 16 * DD) {
            int m = tid >> 3, d = tid & 7;
            sA0[m * K0P + d] = (__bf16)sXf[m * DD + d];
        }
        __syncthreads();                                        // (d)

        // issue async prefetch of x_{t+2} (readers blocked until after (b,t+1))
        if (t + 2 < TT && wave == 0) {
            unsigned long long base = (unsigned long long)(const void*)x
                                    + (unsigned long long)(t + 2) * (DD * 4);
            asm volatile("global_load_async_to_lds_b128 %0, %1, %2"
                         :: "v"(ldsa), "v"(voff), "s"(base) : "memory");
        }

        // ---- layer 1 GEMM: gates1 = [h0_t|h1|1] @ W1 (4 k-chunks) ----
        {
            v8f acc = {};
#pragma unroll
            for (int c = 0; c < 4; ++c) {
                v16bf a = load_afrag(sA1, K1P, c * 32, lane);
                v16bf b = load_bfrag(sW + (size_t)(W0_FRAGS + wave * 4 + c) * FRAG_ELEMS, lane);
                acc = __builtin_amdgcn_wmma_f32_16x16x32_bf16(
                        false, a, false, b, (short)0, acc, false, false);
            }
#pragma unroll
            for (int v = 0; v < 8; ++v) sG1[(rb + v) * GP + col] = acc[v];
        }
        __syncthreads();                                        // (f)

        // ---- layer 1 pointwise (no trailing barrier needed) ----
        {
            float h = lstm_cell(g1e0, sC1[tid], &sC1[tid]);
            *hR1e0 = (__bf16)h;
            if (has1) {
                float h2 = lstm_cell(g1e1, sC1[i1], &sC1[i1]);
                *hR1e1 = (__bf16)h2;
            }
        }
    }
    __syncthreads();

    // ---- final projection: out[m] = h1_last . Wlin + blin ----
    if (tid < 16) {
        float s = 0.0f;
        for (int j = 0; j < HH; ++j)
            s += (float)sA1[tid * K1P + HH + j] * Wlin[j];
        out[row0 + tid] = s + blin[0];
    }
}

// ---------------- launch ----------------
extern "C" void kernel_launch(void* const* d_in, const int* in_sizes, int n_in,
                              void* d_out, int out_size, void* d_ws, size_t ws_size,
                              hipStream_t stream) {
    const float* x    = (const float*)d_in[0];
    const float* Wih0 = (const float*)d_in[1];
    const float* Whh0 = (const float*)d_in[2];
    const float* bih0 = (const float*)d_in[3];
    const float* bhh0 = (const float*)d_in[4];
    const float* Wih1 = (const float*)d_in[5];
    const float* Whh1 = (const float*)d_in[6];
    const float* bih1 = (const float*)d_in[7];
    const float* bhh1 = (const float*)d_in[8];
    const float* Wlin = (const float*)d_in[9];
    const float* blin = (const float*)d_in[10];

    __bf16* wpack = (__bf16*)d_ws;

    pack_weights<<<(WPACK_ELEMS + 255) / 256, 256, 0, stream>>>(
        Wih0, Whh0, bih0, bhh0, Wih1, Whh1, bih1, bhh1, wpack);

    lstm_fused<<<2048 / 16, NTHREADS, SMEM_TOT, stream>>>(
        x, wpack, Wlin, blin, (float*)d_out);
}
